// RobustFill_66151086293512
// MI455X (gfx1250) — matrix operand order repaired and checked by reading, not compiled, not run on hardware
//
#include <hip/hip_runtime.h>

#define HSZ   512
#define BATCH 1024
#define TIN   48
#define TOUT  32
#define G4    2048   // 4*HSZ

typedef __bf16 bf16;
typedef float v4f  __attribute__((ext_vector_type(4)));
typedef float v8f  __attribute__((ext_vector_type(8)));
typedef __bf16 v16bf __attribute__((ext_vector_type(16)));
typedef __bf16 v8bf  __attribute__((ext_vector_type(8)));

__device__ __forceinline__ float sigmoidf_(float x) { return 1.f / (1.f + __expf(-x)); }

// ---------------------------------------------------------------------------
// Pack a row-major f32 weight W[N x ldw] (using columns [col0, col0+K)) into
// bf16 WMMA B fragments (contiguous k order, same as the A staging):
//   out[((kt*Ntiles + nt)*32 + lane)*16 + e] = W[nt*16 + lane%16][col0 + kt*32 + (lane/16)*16 + e]
// Rows n >= N are zero padded (Npad is a multiple of 64).
// ---------------------------------------------------------------------------
__global__ void pack_b_kernel(const float* __restrict__ W, int ldw, int col0,
                              int N, int K, int Npad, bf16* __restrict__ out) {
    int tid = blockIdx.x * blockDim.x + threadIdx.x;
    int total = (K >> 5) * (Npad >> 4) * 32;
    if (tid >= total) return;
    int lane = tid & 31;
    int grp  = tid >> 5;
    int Ntiles = Npad >> 4;
    int nt = grp % Ntiles;
    int kt = grp / Ntiles;
    int n = nt * 16 + (lane & 15);
    int kbase = kt * 32 + ((lane >> 4) << 4);
    bf16* o = out + (size_t)tid * 16;
#pragma unroll
    for (int e = 0; e < 16; ++e) {
        float v = 0.f;
        if (n < N) v = W[(size_t)n * ldw + col0 + kbase + e];
        o[e] = (bf16)v;
    }
}

// ---------------------------------------------------------------------------
// C[M x ldc] (f32) = (accumulate ? C : 0) + A[M x lda] (f32) * B (packed bf16)
// M = 1024 fixed via grid.y = 8 (128 rows/block). Npad multiple of 64,
// K multiple of 32. Workgroup = 8 waves; wave computes 32x32 (2x2 fragments).
// A block staged to LDS with f32->bf16 conversion in A-fragment layout.
// ---------------------------------------------------------------------------
__global__ void __launch_bounds__(256)
gemm_bf16_wmma(const float* __restrict__ A, int lda,
               const bf16* __restrict__ Bp,
               float* __restrict__ C, int ldc,
               int K, int NtilesPad, int accumulate) {
    __shared__ __align__(32) bf16 aT[8][32][16];   // 8 KB

    const int tid  = threadIdx.x;
    const int lane = tid & 31;
    const int wave = tid >> 5;        // 0..7
    const int wm   = wave >> 1;       // 0..3 (M)
    const int wn   = wave & 1;        // 0..1 (N)
    const int br   = blockIdx.y;      // M blocks of 128
    const int bc   = blockIdx.x;      // N blocks of 64
    const int mloc = lane & 15;
    const int half = lane >> 4;

    // A staging: thread handles m-subtile = wave, stage-lane = lane
    const int arow = br * 128 + wave * 16 + mloc;
    const float* abase = A + (size_t)arow * lda + half * 16;

    // B fragments for this wave
    const int nt0 = bc * 4 + wn * 2;
    const bf16* bbase = Bp + ((size_t)nt0 * 32 + lane) * 16;
    const size_t bKstride = (size_t)NtilesPad * 32 * 16;
    const size_t bTileStride = 32 * 16;

    // C addressing: element r of v8f -> row (m0 + 8*half + r), col (n0 + mloc)
    const int m0 = br * 128 + wm * 32;
    const int n0 = bc * 64 + wn * 32;
    float* cbase = C + (size_t)(m0 + 8 * half) * ldc + n0 + mloc;

    v8f acc00, acc01, acc10, acc11;
    if (accumulate) {
#pragma unroll
        for (int r = 0; r < 8; ++r) {
            acc00[r] = cbase[(size_t)r * ldc];
            acc01[r] = cbase[(size_t)r * ldc + 16];
            acc10[r] = cbase[(size_t)(r + 16) * ldc];
            acc11[r] = cbase[(size_t)(r + 16) * ldc + 16];
        }
    } else {
#pragma unroll
        for (int r = 0; r < 8; ++r) { acc00[r] = 0.f; acc01[r] = 0.f; acc10[r] = 0.f; acc11[r] = 0.f; }
    }

    const int Kt = K >> 5;
    for (int kt = 0; kt < Kt; ++kt) {
        // ---- stage A 128x32 block into LDS as bf16 fragments ----
        const float* ap = abase + kt * 32;
        v4f a0 = *(const v4f*)(ap + 0);
        v4f a1 = *(const v4f*)(ap + 4);
        v4f a2 = *(const v4f*)(ap + 8);
        v4f a3 = *(const v4f*)(ap + 12);
        bf16* dst = &aT[wave][lane][0];
#pragma unroll
        for (int i = 0; i < 4; ++i) {
            dst[i]      = (bf16)a0[i];
            dst[4 + i]  = (bf16)a1[i];
            dst[8 + i]  = (bf16)a2[i];
            dst[12 + i] = (bf16)a3[i];
        }
        __syncthreads();

        v16bf af0 = *(const v16bf*)&aT[wm * 2 + 0][lane][0];
        v16bf af1 = *(const v16bf*)&aT[wm * 2 + 1][lane][0];

        const bf16* bp = bbase + (size_t)kt * bKstride;
        v16bf bf0 = *(const v16bf*)(bp);
        v16bf bf1 = *(const v16bf*)(bp + bTileStride);
        if (kt + 1 < Kt) {
            __builtin_prefetch(bp + bKstride, 0, 3);
            __builtin_prefetch(bp + bKstride + bTileStride, 0, 3);
        }

        acc00 = __builtin_amdgcn_wmma_f32_16x16x32_bf16(false, af0, false, bf0, (short)0, acc00, false, false);
        acc01 = __builtin_amdgcn_wmma_f32_16x16x32_bf16(false, af0, false, bf1, (short)0, acc01, false, false);
        acc10 = __builtin_amdgcn_wmma_f32_16x16x32_bf16(false, af1, false, bf0, (short)0, acc10, false, false);
        acc11 = __builtin_amdgcn_wmma_f32_16x16x32_bf16(false, af1, false, bf1, (short)0, acc11, false, false);
        __syncthreads();
    }

#pragma unroll
    for (int r = 0; r < 8; ++r) {
        cbase[(size_t)r * ldc]             = acc00[r];
        cbase[(size_t)r * ldc + 16]        = acc01[r];
        cbase[(size_t)(r + 16) * ldc]      = acc10[r];
        cbase[(size_t)(r + 16) * ldc + 16] = acc11[r];
    }
}

// ---------------------------------------------------------------------------
// scores[t][b] = dot(Henc[t][b][:], u[b][:]) over 512 (Henc bf16, u f32).
// One wave per (t,b).
// ---------------------------------------------------------------------------
__global__ void attn_scores_kernel(const bf16* __restrict__ Henc,
                                   const float* __restrict__ u,
                                   float* __restrict__ scores, int T) {
    int lane = threadIdx.x & 31;
    int gw = (blockIdx.x * blockDim.x + threadIdx.x) >> 5;
    int t = gw >> 10;
    int b = gw & 1023;
    if (t >= T) return;
    const bf16* hp = Henc + ((size_t)t * BATCH + b) * HSZ + lane * 16;
    const float* up = u + (size_t)b * HSZ + lane * 16;
    v8bf h0 = *(const v8bf*)(hp);
    v8bf h1 = *(const v8bf*)(hp + 8);
    float s = 0.f;
#pragma unroll
    for (int i = 0; i < 8; ++i) s += (float)h0[i] * up[i];
#pragma unroll
    for (int i = 0; i < 8; ++i) s += (float)h1[i] * up[8 + i];
    for (int off = 16; off; off >>= 1) s += __shfl_down(s, off, 32);
    if (lane == 0) scores[t * BATCH + b] = s;
}

// ---------------------------------------------------------------------------
// Softmax over t of scores[:,b], then ctx[b][h] = sum_t w[t]*Henc[t][b][h].
// One block (256 threads) per batch element.
// ---------------------------------------------------------------------------
__global__ void attn_ctx_kernel(const float* __restrict__ scores,
                                const bf16* __restrict__ Henc,
                                float* __restrict__ ctx, int ldctx, int T) {
    __shared__ float w[64];
    int b = blockIdx.x;
    if (threadIdx.x == 0) {
        float m = -1e30f;
        for (int t = 0; t < T; ++t) { float v = scores[t * BATCH + b]; m = fmaxf(m, v); }
        float s = 0.f;
        for (int t = 0; t < T; ++t) { float e = __expf(scores[t * BATCH + b] - m); w[t] = e; s += e; }
        float inv = 1.f / s;
        for (int t = 0; t < T; ++t) w[t] *= inv;
    }
    __syncthreads();
    const bf16* hb = Henc + (size_t)b * HSZ;
    for (int hh = threadIdx.x; hh < HSZ; hh += 256) {
        float acc = 0.f;
        const bf16* hp = hb + hh;
        for (int t = 0; t < T; ++t) acc += w[t] * (float)hp[(size_t)t * BATCH * HSZ];
        ctx[(size_t)b * ldctx + hh] = acc;
    }
}

// ---------------------------------------------------------------------------
// LSTM pointwise: gates already hold h@Whh^T (+ ctx@Wctx^T). Add one-hot
// column gather Wih[:, tok] + bih + bhh, apply cell math. Optionally emit
// h to a bf16 Henc time-slot.
// ---------------------------------------------------------------------------
__global__ void lstm_pointwise_kernel(const float* __restrict__ gates,
                                      const float* __restrict__ Wih, int ldwih,
                                      const int* __restrict__ tokens,
                                      const float* __restrict__ bih,
                                      const float* __restrict__ bhh,
                                      float* __restrict__ h, float* __restrict__ c,
                                      bf16* __restrict__ henc_slot) {
    int idx = blockIdx.x * blockDim.x + threadIdx.x;   // BATCH*HSZ
    int b = idx >> 9;
    int j = idx & 511;
    int tok = tokens[b];
    const float* g = gates + (size_t)b * G4;
    float gi = g[j]          + Wih[(size_t)(j)        * ldwih + tok] + bih[j]          + bhh[j];
    float gf = g[512 + j]    + Wih[(size_t)(512 + j)  * ldwih + tok] + bih[512 + j]    + bhh[512 + j];
    float gg = g[1024 + j]   + Wih[(size_t)(1024 + j) * ldwih + tok] + bih[1024 + j]   + bhh[1024 + j];
    float go = g[1536 + j]   + Wih[(size_t)(1536 + j) * ldwih + tok] + bih[1536 + j]   + bhh[1536 + j];
    float cn = sigmoidf_(gf) * c[idx] + sigmoidf_(gi) * tanhf(gg);
    float hn = sigmoidf_(go) * tanhf(cn);
    c[idx] = cn;
    h[idx] = hn;
    if (henc_slot) henc_slot[idx] = (bf16)hn;
}

__global__ void bcast_kernel(float* __restrict__ dst, const float* __restrict__ src512) {
    int idx = blockIdx.x * blockDim.x + threadIdx.x;
    dst[idx] = src512[idx & 511];
}

__global__ void zero_kernel(float* __restrict__ p) {
    p[blockIdx.x * blockDim.x + threadIdx.x] = 0.f;
}

__global__ void copy_h_to_hcat_kernel(const float* __restrict__ h, float* __restrict__ hcat) {
    int idx = blockIdx.x * blockDim.x + threadIdx.x;
    int b = idx >> 9, j = idx & 511;
    hcat[(size_t)b * 1024 + j] = h[idx];
}

__global__ void tanh_bias_kernel(float* __restrict__ e, const float* __restrict__ wb) {
    int idx = blockIdx.x * blockDim.x + threadIdx.x;   // BATCH*128
    e[idx] = tanhf(e[idx] + wb[idx & 127]);
}

// ---------------------------------------------------------------------------
// Per-batch log_softmax over 129 logits (+V_b) and accumulate target logp.
// One wave per batch element.
// ---------------------------------------------------------------------------
__global__ void dec_score_kernel(const float* __restrict__ logits,   // [B][192]
                                 const float* __restrict__ vb,       // [129]
                                 const int* __restrict__ tgt,
                                 float* __restrict__ out) {
    int lane = threadIdx.x & 31;
    int b = (blockIdx.x * blockDim.x + threadIdx.x) >> 5;
    const float* lp = logits + (size_t)b * 192;
    float vals[5];
    float m = -1e30f;
#pragma unroll
    for (int k = 0; k < 5; ++k) {
        int v = lane + 32 * k;
        vals[k] = (v < 129) ? (lp[v] + vb[v]) : -1e30f;
        m = fmaxf(m, vals[k]);
    }
    for (int x = 16; x; x >>= 1) m = fmaxf(m, __shfl_xor(m, x, 32));
    float s = 0.f;
#pragma unroll
    for (int k = 0; k < 5; ++k) s += __expf(vals[k] - m);
    for (int x = 16; x; x >>= 1) s += __shfl_xor(s, x, 32);
    if (lane == 0) {
        int tv = tgt[b];
        out[b] += (lp[tv] + vb[tv]) - (m + __logf(s));
    }
}

// ---------------------------------------------------------------------------
extern "C" void kernel_launch(void* const* d_in, const int* in_sizes, int n_in,
                              void* d_out, int out_size, void* d_ws, size_t ws_size,
                              hipStream_t stream) {
    (void)in_sizes; (void)n_in; (void)out_size; (void)ws_size;

    const int*   input0 = (const int*)d_in[0];
    const int*   input1 = (const int*)d_in[1];
    const int*   target = (const int*)d_in[2];
    const float* Wih0 = (const float*)d_in[3];
    const float* Whh0 = (const float*)d_in[4];
    const float* bih0 = (const float*)d_in[5];
    const float* bhh0 = (const float*)d_in[6];
    const float* Wih1 = (const float*)d_in[7];
    const float* Whh1 = (const float*)d_in[8];
    const float* bih1 = (const float*)d_in[9];
    const float* bhh1 = (const float*)d_in[10];
    const float* Wihd = (const float*)d_in[11];
    const float* Whhd = (const float*)d_in[12];
    const float* bihd = (const float*)d_in[13];
    const float* bhhd = (const float*)d_in[14];
    const float* enc_init_h  = (const float*)d_in[15];
    const float* enc_init_c0 = (const float*)d_in[16];
    const float* enc_init_c1 = (const float*)d_in[17];
    const float* dec_init_c  = (const float*)d_in[18];
    const float* A0  = (const float*)d_in[19];
    const float* A1  = (const float*)d_in[20];
    const float* W_w = (const float*)d_in[21];
    const float* W_b = (const float*)d_in[22];
    const float* V_w = (const float*)d_in[23];
    const float* V_b = (const float*)d_in[24];
    float* out = (float*)d_out;

    char* wp = (char*)d_ws;
    auto carve = [&](size_t bytes) -> char* {
        char* r = wp;
        wp += (bytes + 255) & ~(size_t)255;
        return r;
    };
    bf16*  H0   = (bf16*) carve((size_t)TIN * BATCH * HSZ * 2);
    bf16*  H1   = (bf16*) carve((size_t)TIN * BATCH * HSZ * 2);
    float* gates= (float*)carve((size_t)BATCH * G4 * 4);
    float* u    = (float*)carve((size_t)BATCH * HSZ * 4);
    float* sc   = (float*)carve((size_t)TIN * BATCH * 4);
    float* ctx  = (float*)carve((size_t)BATCH * HSZ * 4);
    float* h    = (float*)carve((size_t)BATCH * HSZ * 4);
    float* c    = (float*)carve((size_t)BATCH * HSZ * 4);
    float* hcat = (float*)carve((size_t)BATCH * 1024 * 4);
    float* emb  = (float*)carve((size_t)BATCH * 128 * 4);
    float* logits = (float*)carve((size_t)BATCH * 192 * 4);
    bf16* Whh0p = (bf16*)carve((size_t)512 * 2048 * 2);
    bf16* Whh1p = (bf16*)carve((size_t)512 * 2048 * 2);
    bf16* Wctxp = (bf16*)carve((size_t)512 * 2048 * 2);
    bf16* Whhdp = (bf16*)carve((size_t)512 * 2048 * 2);
    bf16* A0p   = (bf16*)carve((size_t)512 * 512 * 2);
    bf16* A1p   = (bf16*)carve((size_t)512 * 512 * 2);
    bf16* Wwp   = (bf16*)carve((size_t)1024 * 128 * 2);
    bf16* Vwp   = (bf16*)carve((size_t)128 * 192 * 2);

    auto pack = [&](const float* W, int ldw, int col0, int N, int K, int Npad, bf16* o) {
        int total = (K >> 5) * (Npad >> 4) * 32;
        pack_b_kernel<<<(total + 255) / 256, 256, 0, stream>>>(W, ldw, col0, N, K, Npad, o);
    };
    pack(Whh0, 512,   0, 2048, 512, 2048, Whh0p);
    pack(Whh1, 512,   0, 2048, 512, 2048, Whh1p);
    pack(Wih1, 641, 129, 2048, 512, 2048, Wctxp);   // ctx columns of Wih1
    pack(Whhd, 512,   0, 2048, 512, 2048, Whhdp);
    pack(A0,   512,   0,  512, 512,  512, A0p);
    pack(A1,   512,   0,  512, 512,  512, A1p);
    pack(W_w, 1024,   0,  128, 1024, 128, Wwp);
    pack(V_w,  128,   0,  129, 128,  192, Vwp);

    auto gemm = [&](const float* Ap, int lda, const bf16* Bp, float* Cp, int ldc,
                    int K, int Npad, int accum) {
        dim3 g(Npad >> 6, BATCH >> 7);
        gemm_bf16_wmma<<<g, 256, 0, stream>>>(Ap, lda, Bp, Cp, ldc, K, Npad >> 4, accum);
    };

    zero_kernel<<<4, 256, 0, stream>>>(out);

    // ---------------- encoder 0 ----------------
    bcast_kernel<<<2048, 256, 0, stream>>>(h, enc_init_h);
    bcast_kernel<<<2048, 256, 0, stream>>>(c, enc_init_c0);
    for (int t = 0; t < TIN; ++t) {
        gemm(h, 512, Whh0p, gates, 2048, 512, 2048, 0);
        lstm_pointwise_kernel<<<2048, 256, 0, stream>>>(
            gates, Wih0, 129, input0 + t * BATCH, bih0, bhh0, h, c,
            H0 + (size_t)t * BATCH * HSZ);
    }

    // ---------------- encoder 1 (attends to H0) ----------------
    bcast_kernel<<<2048, 256, 0, stream>>>(h, enc_init_h);
    bcast_kernel<<<2048, 256, 0, stream>>>(c, enc_init_c1);
    for (int t = 0; t < TIN; ++t) {
        gemm(h, 512, A0p, u, 512, 512, 512, 0);
        attn_scores_kernel<<<TIN * 128, 256, 0, stream>>>(H0, u, sc, TIN);
        attn_ctx_kernel<<<BATCH, 256, 0, stream>>>(sc, H0, ctx, 512, TIN);
        gemm(h,   512, Whh1p, gates, 2048, 512, 2048, 0);
        gemm(ctx, 512, Wctxp, gates, 2048, 512, 2048, 1);
        lstm_pointwise_kernel<<<2048, 256, 0, stream>>>(
            gates, Wih1, 641, input1 + t * BATCH, bih1, bhh1, h, c,
            H1 + (size_t)t * BATCH * HSZ);
    }

    // ---------------- decoder (attends to H1) ----------------
    bcast_kernel<<<2048, 256, 0, stream>>>(c, dec_init_c);   // h keeps enc1 final state
    for (int t = 0; t < TOUT; ++t) {
        gemm(h, 512, A1p, u, 512, 512, 512, 0);
        attn_scores_kernel<<<TIN * 128, 256, 0, stream>>>(H1, u, sc, TIN);
        attn_ctx_kernel<<<BATCH, 256, 0, stream>>>(sc, H1, hcat + 512, 1024, TIN);
        copy_h_to_hcat_kernel<<<2048, 256, 0, stream>>>(h, hcat);
        gemm(hcat, 1024, Wwp, emb, 128, 1024, 128, 0);
        tanh_bias_kernel<<<512, 256, 0, stream>>>(emb, W_b);
        gemm(emb, 128, Vwp, logits, 192, 128, 192, 0);
        dec_score_kernel<<<128, 256, 0, stream>>>(logits, V_b, target + t * BATCH, out);
        gemm(h, 512, Whhdp, gates, 2048, 512, 2048, 0);
        lstm_pointwise_kernel<<<2048, 256, 0, stream>>>(
            gates, Wihd, 129, target + t * BATCH, bihd, bhhd, h, c, nullptr);
    }
}
